// LSTM_89953795048070
// MI455X (gfx1250) — matrix operand order repaired and checked by reading, not compiled, not run on hardware
//
#include <hip/hip_runtime.h>
#include <cstdint>

// ---------------------------------------------------------------------------
// LSTM-like scan, MI455X (gfx1250, wave32, WMMA).
//   Per step: pre[g] = x_h @ W_g^T + b_g  (4 gates), fused epilogue:
//     g=tanh, i/f/o=sigmoid, new_c = g*i + h*f, new_h = tanh(new_c)*o
//     carry: h <- new_c (f32), c <- new_h (bf16 for next GEMM A)
//   GEMM: bf16 WMMA 16x16x32, f32 accum. Weights packed once into a contiguous-k (k=16h+s)
//   B-matrix register striping; A staged to LDS via async-load-to-LDS.
// Workspace: Wpk 10.5MB | hstate 4MB | cactA 2MB | cactB 2MB  (~18.9MB)
// ---------------------------------------------------------------------------

#define SEQ_LEN    128
#define INPUT_DIM  256
#define HIDDEN_DIM 1024
#define NUM_CLS    10
#define BATCH      1024
#define KTOT       (INPUT_DIM + HIDDEN_DIM)   // 1280
#define NKT        (KTOT / 32)                // 40 k-tiles
#define NNT        (HIDDEN_DIM / 16)          // 64 n-tiles

typedef __bf16    v16bf __attribute__((ext_vector_type(16)));
typedef float     v8f   __attribute__((ext_vector_type(8)));
typedef float     v4f   __attribute__((ext_vector_type(4)));
typedef uint32_t  v4u   __attribute__((ext_vector_type(4)));
typedef int       v4i_g __attribute__((vector_size(16)));   // matches builtin sig

#define GAS1 __attribute__((address_space(1)))
#define LAS3 __attribute__((address_space(3)))

#if __has_builtin(__builtin_amdgcn_global_load_async_to_lds_b128)
#define USE_ASYNC_LDS 1
#endif

union ABFrag { v4u u[2]; v16bf v; };

__device__ __forceinline__ uint32_t pack2_bf16(float a, float b) {
  union { __bf16 h[2]; uint32_t u; } p;
  p.h[0] = (__bf16)a; p.h[1] = (__bf16)b;
  return p.u;
}

__device__ __forceinline__ float sigm(float x) {
  return 1.0f / (1.0f + __expf(-x));
}

__device__ __forceinline__ v8f wmma_bf16(v16bf a, v16bf b, v8f c) {
  // D = A(16x32) * B(32x16) + C, f32 accum
  return __builtin_amdgcn_wmma_f32_16x16x32_bf16(
      false, a, false, b, (short)0, c, false, false);
}

#if defined(USE_ASYNC_LDS)
// 16B global -> LDS async copy (tracked by ASYNCcnt).
__device__ __forceinline__ void async_copy16(const void* gsrc, void* ldst) {
  __builtin_amdgcn_global_load_async_to_lds_b128(
      (GAS1 v4i_g*)(uintptr_t)gsrc,   // flat addr == global addr for AS1
      (LAS3 v4i_g*)ldst,              // addrspacecast flat -> LDS
      0, 0);
}
#endif

// ---------------------------------------------------------------------------
// Pack w_{g,i,f,o} [H, KTOT] f32 -> bf16 in a contiguous-k WMMA B layout:
// elem layout: (((gate*64 + ntile)*40 + ktile)*32 + lane)*16
//   lane: N = ntile*16 + (lane&15), khalf = lane>>4
//   elem j: K = ktile*32 + khalf*16 + j   (B[K][N] = w[N][K])
// ---------------------------------------------------------------------------
__global__ void pack_weights_kernel(const float* __restrict__ wg,
                                    const float* __restrict__ wi,
                                    const float* __restrict__ wf,
                                    const float* __restrict__ wo,
                                    uint16_t* __restrict__ Wpk) {
  int tid = blockIdx.x * 256 + threadIdx.x;
  if (tid >= 4 * NNT * NKT * 32) return;
  int lane = tid & 31;
  int tmp  = tid >> 5;
  int kt   = tmp % NKT;
  int nt   = (tmp / NKT) % NNT;
  int g    = tmp / (NKT * NNT);
  const float* w = (g == 0) ? wg : (g == 1) ? wi : (g == 2) ? wf : wo;
  int n  = nt * 16 + (lane & 15);
  int k0 = kt * 32 + (lane >> 4) * 16;
  const float* src = w + (size_t)n * KTOT + k0;   // 16 contiguous f32

  v4f f0 = *(const v4f*)(src + 0);
  v4f f1 = *(const v4f*)(src + 4);
  v4f f2 = *(const v4f*)(src + 8);
  v4f f3 = *(const v4f*)(src + 12);
  v4u o0, o1;
  o0.x = pack2_bf16(f0.x, f0.y); o0.y = pack2_bf16(f0.z, f0.w);
  o0.z = pack2_bf16(f1.x, f1.y); o0.w = pack2_bf16(f1.z, f1.w);
  o1.x = pack2_bf16(f2.x, f2.y); o1.y = pack2_bf16(f2.z, f2.w);
  o1.z = pack2_bf16(f3.x, f3.y); o1.w = pack2_bf16(f3.z, f3.w);
  uint16_t* dst = Wpk + (size_t)tid * 16;
  *(v4u*)(dst + 0) = o0;
  *(v4u*)(dst + 8) = o1;
}

__global__ void init_state_kernel(float* __restrict__ hstate,
                                  uint16_t* __restrict__ cactA) {
  int i = blockIdx.x * 256 + threadIdx.x;
  if (i < BATCH * HIDDEN_DIM) { hstate[i] = 0.0f; cactA[i] = 0; }
}

// ---------------------------------------------------------------------------
// One recurrent step.  Grid (H/32=32, B/128=8), block 256 (8 waves).
// WG tile: M=128 (batch) x N=32 (hidden cols) x 4 gates, K=1280.
// A-tile (128x32 bf16 = 8KB) double-buffered in LDS, stored as [row][16 u32]
// (K packed in bf16 pairs).  K-tiles 0..7 gather embeds, 8..39 read cact.
// ---------------------------------------------------------------------------
__global__ void __launch_bounds__(256)
lstm_step_kernel(const int* __restrict__ x, const float* __restrict__ embed_w,
                 const uint16_t* __restrict__ Wpk,
                 const float* __restrict__ b_g, const float* __restrict__ b_i,
                 const float* __restrict__ b_f, const float* __restrict__ b_o,
                 const uint16_t* __restrict__ cact_in,
                 uint16_t* __restrict__ cact_out,
                 float* __restrict__ hstate, int t) {
  __shared__ __align__(16) uint32_t Atile[2][128 * 16];

  const int tid  = threadIdx.x;
  const int lane = tid & 31;
  const int wave = tid >> 5;
  const int wm   = wave & 3;    // 4 waves along M
  const int wn   = wave >> 2;   // 2 waves along N
  const int mWG  = blockIdx.y * 128;
  const int mW   = mWG + wm * 32;
  const int nW   = blockIdx.x * 32 + wn * 16;
  const int ntW  = blockIdx.x * 2 + wn;

  // Per-gate B fragment bases (v4u units; 32B/lane/ktile)
  const v4u* bbase[4];
#pragma unroll
  for (int g = 0; g < 4; ++g)
    bbase[g] = (const v4u*)Wpk +
               ((((size_t)g * NNT + ntW) * NKT) * 32 + lane) * 2;

  v8f acc[2][4];
#pragma unroll
  for (int s = 0; s < 2; ++s)
#pragma unroll
    for (int g = 0; g < 4; ++g) acc[s][g] = (v8f)0.0f;

  // ---- A-tile staging (512 16B chunks, 2 per thread) ----
  auto stage = [&](int buf, int kt) {
#pragma unroll
    for (int c = tid; c < 512; c += 256) {
      int r = c >> 2, c4 = c & 3;
      uint32_t* ldst = &Atile[buf][r * 16 + c4 * 4];
      if (kt < INPUT_DIM / 32) {                // embedding gather + cvt
        int b   = mWG + r;
        int idx = x[b * SEQ_LEN + t];
        const float* src = embed_w + (size_t)idx * INPUT_DIM + kt * 32 + c4 * 8;
        v4f f0 = *(const v4f*)(src + 0);
        v4f f1 = *(const v4f*)(src + 4);
        v4u o;
        o.x = pack2_bf16(f0.x, f0.y); o.y = pack2_bf16(f0.z, f0.w);
        o.z = pack2_bf16(f1.x, f1.y); o.w = pack2_bf16(f1.z, f1.w);
        *(v4u*)ldst = o;
      } else {                                  // recurrent state, bf16
        const uint16_t* src =
            cact_in + (size_t)(mWG + r) * HIDDEN_DIM + (kt * 32 - INPUT_DIM) + c4 * 8;
#if defined(USE_ASYNC_LDS)
        async_copy16(src, ldst);
#else
        *(v4u*)ldst = *(const v4u*)src;
#endif
      }
    }
  };

  stage(0, 0);

  for (int kt = 0; kt < NKT; ++kt) {
    const int buf = kt & 1;
#if defined(USE_ASYNC_LDS)
#if __has_builtin(__builtin_amdgcn_s_wait_asynccnt)
    __builtin_amdgcn_s_wait_asynccnt(0);
#else
    asm volatile("s_wait_asynccnt 0" ::: "memory");
#endif
#endif
    __syncthreads();
    if (kt + 1 < NKT) stage(buf ^ 1, kt + 1);

    // B fragments (global, L2-resident packed weights): 32B/lane/gate
    ABFrag bf[4];
#pragma unroll
    for (int g = 0; g < 4; ++g) {
      const v4u* p = bbase[g] + (size_t)kt * 64;
      bf[g].u[0] = p[0];
      bf[g].u[1] = p[1];
    }
    // A fragments from LDS (contiguous k = 16h + s, same order as packed B)
    ABFrag af[2];
#pragma unroll
    for (int s = 0; s < 2; ++s) {
      const uint32_t* row = &Atile[buf][(wm * 32 + s * 16 + (lane & 15)) * 16];
      int h = lane >> 4;
      af[s].u[0] = *(const v4u*)(row + h * 8);
      af[s].u[1] = *(const v4u*)(row + h * 8 + 4);
    }
#pragma unroll
    for (int g = 0; g < 4; ++g)
#pragma unroll
      for (int s = 0; s < 2; ++s)
        acc[s][g] = wmma_bf16(af[s].v, bf[g].v, acc[s][g]);
  }

  // ---- fused gate epilogue ----
  const int n = nW + (lane & 15);
  const float bg = b_g[n], bi = b_i[n], bff = b_f[n], bo = b_o[n];
#pragma unroll
  for (int s = 0; s < 2; ++s) {
    const int m0 = mW + s * 16 + 8 * (lane >> 4);
#pragma unroll
    for (int j = 0; j < 8; ++j) {
      const int b = m0 + j;
      float g = tanhf(acc[s][0][j] + bg);
      float i = sigm(acc[s][1][j] + bi);
      float f = sigm(acc[s][2][j] + bff);
      float o = sigm(acc[s][3][j] + bo);
      const size_t off = (size_t)b * HIDDEN_DIM + n;
      float h_old = hstate[off];
      float new_c = g * i + h_old * f;
      float new_h = tanhf(new_c) * o;
      hstate[off] = new_c;                       // carry 'h' <- new_c
      union { __bf16 h; uint16_t u; } cv;
      cv.h = (__bf16)new_h;                      // carry 'c' <- new_h (bf16)
      cact_out[off] = cv.u;
    }
  }
}

// ---------------------------------------------------------------------------
// out[b,:] = softmax(h[b,:] @ w_p^T + b_p) ; one wave per batch row.
// ---------------------------------------------------------------------------
__global__ void proj_softmax_kernel(const float* __restrict__ hstate,
                                    const float* __restrict__ wp,
                                    const float* __restrict__ bp,
                                    float* __restrict__ out) {
  int wid  = (blockIdx.x * blockDim.x + threadIdx.x) >> 5;
  int lane = threadIdx.x & 31;
  if (wid >= BATCH) return;
  const float* hrow = hstate + (size_t)wid * HIDDEN_DIM;
  float logits[NUM_CLS];
#pragma unroll
  for (int c = 0; c < NUM_CLS; ++c) {
    float s = 0.0f;
    for (int k = lane; k < HIDDEN_DIM; k += 32)
      s += hrow[k] * wp[(size_t)c * HIDDEN_DIM + k];
#pragma unroll
    for (int off = 16; off > 0; off >>= 1) s += __shfl_xor(s, off, 32);
    logits[c] = s + bp[c];
  }
  float mx = logits[0];
#pragma unroll
  for (int c = 1; c < NUM_CLS; ++c) mx = fmaxf(mx, logits[c]);
  float den = 0.0f;
#pragma unroll
  for (int c = 0; c < NUM_CLS; ++c) { logits[c] = __expf(logits[c] - mx); den += logits[c]; }
  if (lane == 0) {
    float inv = 1.0f / den;
#pragma unroll
    for (int c = 0; c < NUM_CLS; ++c) out[(size_t)wid * NUM_CLS + c] = logits[c] * inv;
  }
}

// ---------------------------------------------------------------------------
extern "C" void kernel_launch(void* const* d_in, const int* in_sizes, int n_in,
                              void* d_out, int out_size, void* d_ws, size_t ws_size,
                              hipStream_t stream) {
  const int*   x       = (const int*)d_in[0];
  const float* embed_w = (const float*)d_in[1];
  const float* w_g     = (const float*)d_in[2];
  const float* w_i     = (const float*)d_in[3];
  const float* w_f     = (const float*)d_in[4];
  const float* w_o     = (const float*)d_in[5];
  const float* b_g     = (const float*)d_in[6];
  const float* b_i     = (const float*)d_in[7];
  const float* b_f     = (const float*)d_in[8];
  const float* b_o     = (const float*)d_in[9];
  const float* w_p     = (const float*)d_in[10];
  const float* b_p     = (const float*)d_in[11];
  float* out = (float*)d_out;

  char* ws = (char*)d_ws;
  const size_t wpk_bytes   = (size_t)4 * HIDDEN_DIM * KTOT * 2;      // 10,485,760
  const size_t hst_bytes   = (size_t)BATCH * HIDDEN_DIM * 4;         //  4,194,304
  const size_t cact_bytes  = (size_t)BATCH * HIDDEN_DIM * 2;         //  2,097,152
  uint16_t* Wpk    = (uint16_t*)(ws);
  float*    hstate = (float*)(ws + wpk_bytes);
  uint16_t* cactA  = (uint16_t*)(ws + wpk_bytes + hst_bytes);
  uint16_t* cactB  = (uint16_t*)(ws + wpk_bytes + hst_bytes + cact_bytes);

  pack_weights_kernel<<<(4 * NNT * NKT * 32 + 255) / 256, 256, 0, stream>>>(
      w_g, w_i, w_f, w_o, Wpk);
  init_state_kernel<<<(BATCH * HIDDEN_DIM + 255) / 256, 256, 0, stream>>>(
      hstate, cactA);

  for (int t = 0; t < SEQ_LEN; ++t) {
    const uint16_t* cin = (t & 1) ? cactB : cactA;
    uint16_t*      cout = (t & 1) ? cactA : cactB;
    lstm_step_kernel<<<dim3(HIDDEN_DIM / 32, BATCH / 128), 256, 0, stream>>>(
        x, embed_w, Wpk, b_g, b_i, b_f, b_o, cin, cout, hstate, t);
  }

  proj_softmax_kernel<<<(BATCH * 32) / 256, 256, 0, stream>>>(hstate, w_p, b_p, out);
}
